// HNX_M_V4_90357521973816
// MI455X (gfx1250) — matrix operand order repaired and checked by reading, not compiled, not run on hardware
//
#include <hip/hip_runtime.h>
#include <stddef.h>
#include <stdint.h>

// ---------------------------------------------------------------------------
// Types for CDNA5 WMMA
// ---------------------------------------------------------------------------
typedef __bf16 bf16_t;
typedef __attribute__((ext_vector_type(16))) __bf16 v16bf;
typedef __attribute__((ext_vector_type(8)))  float  v8f;

union Frag16 { v16bf v; uint4 q[2]; };

// Problem dimensions (fixed by the reference)
#define DIM_B   4
#define DIM_T   2048
#define DIM_DI  1024
#define DIM_H   1024
#define DIM_O   1024
#define DIM_S   128
#define NTOK    (DIM_B * DIM_T)     // 8192 tokens
#define TWO_H   (2 * DIM_H)

__device__ __forceinline__ float sigmoidf_(float x) { return 1.0f / (1.0f + __expf(-x)); }
__device__ __forceinline__ float siluf_(float x)    { return x / (1.0f + __expf(-x)); }

// ---- CDNA5 async global->LDS copy (Tensor/async path, tracked by ASYNCcnt) --
__device__ __forceinline__ void async_copy_b128(const void* gaddr, void* lds) {
  // LDS byte offset lives in the low 32 bits of a generic pointer to LDS.
  asm volatile("global_load_async_to_lds_b128 %0, %1, off"
               :: "v"((unsigned)(uintptr_t)lds), "v"(gaddr)
               : "memory");
}
__device__ __forceinline__ void wait_asynccnt0() {
#if __has_builtin(__builtin_amdgcn_s_wait_asynccnt)
  __builtin_amdgcn_s_wait_asynccnt(0);
#else
  asm volatile("s_wait_asynccnt 0" ::: "memory");
#endif
}

// ---------------------------------------------------------------------------
// f32 -> bf16 conversion (row-major, for activations)
// ---------------------------------------------------------------------------
__global__ __launch_bounds__(256) void k_cvt_bf16(const float* __restrict__ in,
                                                  bf16_t* __restrict__ out, size_t n) {
  size_t i = (size_t)blockIdx.x * 256 + threadIdx.x;
  if (i < n) out[i] = (bf16_t)in[i];
}

// f32 [R][C] -> bf16 [C][R] tiled transpose (for weights; all dims % 32 == 0)
__global__ __launch_bounds__(256) void k_cvt_bf16_t(const float* __restrict__ in,
                                                    bf16_t* __restrict__ out,
                                                    int R, int C) {
  __shared__ float tile[32][33];
  int bx = blockIdx.x;               // C / 32
  int by = blockIdx.y;               // R / 32
  int tx = threadIdx.x & 31;
  int ty = threadIdx.x >> 5;         // 0..7
#pragma unroll
  for (int i = 0; i < 32; i += 8)
    tile[ty + i][tx] = in[(size_t)(by * 32 + ty + i) * C + bx * 32 + tx];
  __syncthreads();
#pragma unroll
  for (int i = 0; i < 32; i += 8)
    out[(size_t)(bx * 32 + ty + i) * R + by * 32 + tx] = (bf16_t)tile[tx][ty + i];
}

// decayed memory bank, transposed: out[h][s] = bf16(memory[s][h]*sigmoid(mdec[s]))
__global__ __launch_bounds__(256) void k_membank_t(const float* __restrict__ mem,
                                                   const float* __restrict__ mdec,
                                                   bf16_t* __restrict__ out) {
  __shared__ float tile[32][33];
  int bx = blockIdx.x;               // H/32
  int by = blockIdx.y;               // S/32
  int tx = threadIdx.x & 31;
  int ty = threadIdx.x >> 5;
#pragma unroll
  for (int i = 0; i < 32; i += 8) {
    int s = by * 32 + ty + i;
    tile[ty + i][tx] = mem[(size_t)s * DIM_H + bx * 32 + tx] * sigmoidf_(mdec[s]);
  }
  __syncthreads();
#pragma unroll
  for (int i = 0; i < 32; i += 8)
    out[(size_t)(bx * 32 + ty + i) * DIM_S + by * 32 + tx] = (bf16_t)tile[tx][ty + i];
}

// ---------------------------------------------------------------------------
// bf16 WMMA GEMM (TN):  C[M,N] = A[M,K] @ BT[N,K]^T (+ bias[N])
// Block tile 128x128, BK=32, 256 threads = 8 waves (4 x 2), wave tile 32x64.
// Async global->LDS staging (ASYNCcnt) + LDS double buffering; all fragment
// reads are contiguous ds_load_b128.
// ---------------------------------------------------------------------------
#define BM 128
#define BN 128
#define BKK 32
#define SA 40    // LDS row stride in elements; 80B = 5*16B, keeps b128 alignment

__global__ __launch_bounds__(256) void k_gemm_bf16_tn(const bf16_t* __restrict__ A,
                                                      const bf16_t* __restrict__ BT,
                                                      const float* __restrict__ bias,
                                                      float* __restrict__ C,
                                                      int M, int N, int K) {
  __shared__ __align__(16) bf16_t As[2][BM * SA];   // 2 x 10240 B
  __shared__ __align__(16) bf16_t Bs[2][BN * SA];   // 2 x 10240 B

  const int tid  = threadIdx.x;
  const int lane = tid & 31;
  const int wave = tid >> 5;      // 0..7
  const int wr   = wave & 3;      // 0..3  -> M
  const int wc   = wave >> 2;     // 0..1  -> N
  const int half = (lane >> 4) & 1;
  const int l15  = lane & 15;
  const int bm   = blockIdx.y * BM;
  const int bn   = blockIdx.x * BN;

  // staging decomposition: 512 quads per tile, 2 per thread (rows r, r+64)
  const int srow = tid >> 2;        // 0..63
  const int sseg = tid & 3;         // 0..3 (8 bf16 each)
  const bf16_t* pa0 = A  + (size_t)(bm + srow)      * K + sseg * 8;
  const bf16_t* pa1 = A  + (size_t)(bm + srow + 64) * K + sseg * 8;
  const bf16_t* pb0 = BT + (size_t)(bn + srow)      * K + sseg * 8;
  const bf16_t* pb1 = BT + (size_t)(bn + srow + 64) * K + sseg * 8;

  bf16_t* la0[2]; bf16_t* la1[2]; bf16_t* lb0[2]; bf16_t* lb1[2];
#pragma unroll
  for (int b = 0; b < 2; ++b) {
    la0[b] = &As[b][srow * SA + sseg * 8];
    la1[b] = &As[b][(srow + 64) * SA + sseg * 8];
    lb0[b] = &Bs[b][srow * SA + sseg * 8];
    lb1[b] = &Bs[b][(srow + 64) * SA + sseg * 8];
  }

  v8f acc[2][4] = {};

  // prologue: async-stage tile 0 into buffer 0
  async_copy_b128(pa0, la0[0]);
  async_copy_b128(pa1, la1[0]);
  async_copy_b128(pb0, lb0[0]);
  async_copy_b128(pb1, lb1[0]);
  wait_asynccnt0();
  __syncthreads();

  int cur = 0;
  for (int kk = 0; kk < K; kk += BKK) {
    const bool more = (kk + BKK) < K;
    if (more) {
      // async-stage next K tile into the other buffer; overlaps WMMAs below
      const int nxt = cur ^ 1;
      async_copy_b128(pa0 + kk + BKK, la0[nxt]);
      async_copy_b128(pa1 + kk + BKK, la1[nxt]);
      async_copy_b128(pb0 + kk + BKK, lb0[nxt]);
      async_copy_b128(pb1 + kk + BKK, lb1[nxt]);
    }

    const bf16_t* Ab = As[cur];
    const bf16_t* Bb = Bs[cur];

    // ---- fragments: each is exactly two contiguous b128 LDS reads
    Frag16 af[2];
#pragma unroll
    for (int r = 0; r < 2; ++r) {
      int m = wr * 32 + r * 16 + l15;
      af[r].q[0] = *(const uint4*)(Ab + m * SA + 8 * half);        // K 0..7 / 8..15
      af[r].q[1] = *(const uint4*)(Ab + m * SA + 16 + 8 * half);   // K 16..23 / 24..31
    }
    Frag16 bfr[4];
#pragma unroll
    for (int c = 0; c < 4; ++c) {
      int n = wc * 64 + c * 16 + l15;
      bfr[c].q[0] = *(const uint4*)(Bb + n * SA + 8 * half);       // K 0..7 / 8..15
      bfr[c].q[1] = *(const uint4*)(Bb + n * SA + 16 + 8 * half);  // K 16..23 / 24..31
    }

#pragma unroll
    for (int r = 0; r < 2; ++r)
#pragma unroll
      for (int c = 0; c < 4; ++c)
        acc[r][c] = __builtin_amdgcn_wmma_f32_16x16x32_bf16(
            false, af[r].v, false, bfr[c].v, (short)0, acc[r][c], false, false);

    if (more) {
      wait_asynccnt0();   // own async writes to nxt complete
      __syncthreads();    // everyone's writes visible; everyone's reads of cur done
    }
    cur ^= 1;
  }

  // ---- epilogue: C VGPR i -> M = i + 8*half, N = lane&15 (16-lane row bursts)
#pragma unroll
  for (int r = 0; r < 2; ++r) {
#pragma unroll
    for (int c = 0; c < 4; ++c) {
      int n = bn + wc * 64 + c * 16 + l15;
      float bv = bias ? bias[n] : 0.0f;
#pragma unroll
      for (int i = 0; i < 8; ++i) {
        int m = bm + wr * 32 + r * 16 + i + 8 * half;
        C[(size_t)m * N + n] = acc[r][c][i] + bv;
      }
    }
  }
}

// ---------------------------------------------------------------------------
// torsion split + causal depthwise conv(k=2) + SiLU ->  y[NTOK,H]
// ---------------------------------------------------------------------------
__global__ __launch_bounds__(256) void k_torsion_conv(const float* __restrict__ h,
                                                      const float* __restrict__ dtb,
                                                      const float* __restrict__ ck,
                                                      float* __restrict__ y) {
  size_t idx = (size_t)blockIdx.x * 256 + threadIdx.x;   // over NTOK*H
  int hh = (int)(idx % DIM_H);
  size_t n = idx / DIM_H;
  int t = (int)(n % DIM_T);
  const float* hr = h + n * (size_t)TWO_H;
  float dtv = hr[DIM_H + hh] + dtb[hh];
  float x1  = hr[hh] * siluf_(dtv);
  float xp  = 0.0f;
  if (t > 0) {
    const float* hp = hr - TWO_H;
    float dtp = hp[DIM_H + hh] + dtb[hh];
    xp = hp[hh] * siluf_(dtp);
  }
  float u = ck[hh * 2 + 0] * xp + ck[hh * 2 + 1] * x1;
  y[idx] = siluf_(u);
}

// ---------------------------------------------------------------------------
// forward EMA scan: one lane per (b,h) stream, coalesced across h
// ---------------------------------------------------------------------------
__global__ __launch_bounds__(256) void k_scan_fwd(const float* __restrict__ y,
                                                  const float* __restrict__ decay,
                                                  float* __restrict__ fwd,
                                                  bf16_t* __restrict__ fwdb) {
  int idx = blockIdx.x * 256 + threadIdx.x;   // b*H + h
  int b = idx / DIM_H, hh = idx % DIM_H;
  float d  = sigmoidf_(decay[hh]);
  float om = 1.0f - d;
  float s  = 0.0f;
  size_t base = (size_t)b * DIM_T * DIM_H + hh;
  for (int t = 0; t < DIM_T; ++t) {
    size_t o = base + (size_t)t * DIM_H;
    s = d * s + om * y[o];
    fwd[o]  = s;
    fwdb[o] = (bf16_t)s;
  }
}

// ---------------------------------------------------------------------------
// per-token entropy gate + memory gate (block per token, reductions over H)
// ---------------------------------------------------------------------------
__global__ __launch_bounds__(256) void k_entropy_gates(const float* __restrict__ fwd,
                                                       const float* __restrict__ Wg,
                                                       const float* __restrict__ bmg,
                                                       const float* __restrict__ sw,
                                                       const float* __restrict__ sb,
                                                       float* __restrict__ gate,
                                                       float* __restrict__ gmem) {
  __shared__ float r0[256], r1[256], r2[256];
  int n = blockIdx.x;
  const float* row = fwd + (size_t)n * DIM_H;
  int tid = threadIdx.x;

  float mx = -1e30f;
  for (int hh = tid; hh < DIM_H; hh += 256) mx = fmaxf(mx, row[hh]);
  r0[tid] = mx; __syncthreads();
  for (int s = 128; s > 0; s >>= 1) {
    if (tid < s) r0[tid] = fmaxf(r0[tid], r0[tid + s]);
    __syncthreads();
  }
  mx = r0[0]; __syncthreads();

  float se = 0.0f, sfe = 0.0f, dot = 0.0f;
  for (int hh = tid; hh < DIM_H; hh += 256) {
    float f = row[hh];
    float e = __expf(f - mx);
    se += e; sfe += f * e; dot += f * Wg[hh];
  }
  r0[tid] = se; r1[tid] = sfe; r2[tid] = dot; __syncthreads();
  for (int s = 128; s > 0; s >>= 1) {
    if (tid < s) { r0[tid] += r0[tid + s]; r1[tid] += r1[tid + s]; r2[tid] += r2[tid + s]; }
    __syncthreads();
  }
  if (tid == 0) {
    float lse  = mx + __logf(r0[0]);
    float ent  = lse - r1[0] / r0[0];          // lse - E[f]
    gate[n] = sigmoidf_(sw[0] * ent + sb[0]);
    gmem[n] = sigmoidf_(r2[0] + bmg[0]);
  }
}

// ---------------------------------------------------------------------------
// softmax over S=128 slots, one wave32 per row (4 values per lane)
// ---------------------------------------------------------------------------
__global__ __launch_bounds__(256) void k_softmax128(const float* __restrict__ logits,
                                                    bf16_t* __restrict__ out) {
  int row  = blockIdx.x * 8 + (threadIdx.x >> 5);
  int lane = threadIdx.x & 31;
  const float* p = logits + (size_t)row * DIM_S;
  float v[4];
  float mx = -1e30f;
#pragma unroll
  for (int i = 0; i < 4; ++i) { v[i] = p[lane + 32 * i]; mx = fmaxf(mx, v[i]); }
#pragma unroll
  for (int off = 16; off > 0; off >>= 1) mx = fmaxf(mx, __shfl_xor(mx, off, 32));
  float s = 0.0f;
#pragma unroll
  for (int i = 0; i < 4; ++i) { v[i] = __expf(v[i] - mx); s += v[i]; }
#pragma unroll
  for (int off = 16; off > 0; off >>= 1) s += __shfl_xor(s, off, 32);
  float inv = 1.0f / s;
#pragma unroll
  for (int i = 0; i < 4; ++i) out[(size_t)row * DIM_S + lane + 32 * i] = (bf16_t)(v[i] * inv);
}

// ---------------------------------------------------------------------------
// reverse EMA scan + contrast + gated 3-way fusion -> weighted (bf16)
// ---------------------------------------------------------------------------
__global__ __launch_bounds__(256) void k_scan_bwd_fuse(const float* __restrict__ fwd,
                                                       const float* __restrict__ mem_out,
                                                       const float* __restrict__ gate,
                                                       const float* __restrict__ gmem,
                                                       const float* __restrict__ decay_bwd,
                                                       const float* __restrict__ fw,
                                                       bf16_t* __restrict__ outw) {
  int idx = blockIdx.x * 256 + threadIdx.x;   // b*H + h
  int b = idx / DIM_H, hh = idx % DIM_H;
  float d  = sigmoidf_(decay_bwd[hh]);
  float om = 1.0f - d;
  float f0 = fw[0], f1 = fw[1], f2 = fw[2];
  float s  = 0.0f;
  size_t base = (size_t)b * DIM_T * DIM_H + hh;
  for (int t = DIM_T - 1; t >= 0; --t) {
    size_t o = base + (size_t)t * DIM_H;
    float f = fwd[o];
    s = d * s + om * f;
    float bwd = (f - s) * gate[b * DIM_T + t];
    float w = f0 * f + f1 * bwd + f2 * mem_out[o] * gmem[b * DIM_T + t];
    outw[o] = (bf16_t)w;
  }
}

// ---------------------------------------------------------------------------
// workspace layout (bytes, 256-aligned), with reuse
// ---------------------------------------------------------------------------
#define OFF_H        ((size_t)0)                     // h [NTOK,2H] f32 (64MB); reused: fwd f32
#define OFF_FWD      OFF_H
#define OFF_Y        ((size_t)67108864)              // y [NTOK,H] f32 (32MB); reused: mem_out
#define OFF_MEMOUT   OFF_Y
#define OFF_XB       ((size_t)100663296)             // x bf16 (16MB); reused: weighted bf16
#define OFF_WEIGHTB  OFF_XB
#define OFF_FWDB     ((size_t)117440512)             // fwd bf16 (16MB)
#define OFF_WIB      ((size_t)134217728)             // W_in^T bf16 (4MB); reused: logits f32
#define OFF_LOGITS   OFF_WIB
#define OFF_SWB      ((size_t)138412032)             // sw bf16 (2MB)
#define OFF_WSLOTB   ((size_t)140509184)             // W_slot_bwd^T bf16 (256KB)
#define OFF_MEMBANK  ((size_t)140771328)             // mem bank^T bf16 (256KB)
#define OFF_WOUTB    ((size_t)141033472)             // W_out^T bf16 (2MB)
#define OFF_GATE     ((size_t)143130624)             // gate f32 (32KB)
#define OFF_GMEM     ((size_t)143163392)             // gmem f32 (32KB)

extern "C" void kernel_launch(void* const* d_in, const int* in_sizes, int n_in,
                              void* d_out, int out_size, void* d_ws, size_t ws_size,
                              hipStream_t stream) {
  (void)in_sizes; (void)n_in; (void)out_size; (void)ws_size;

  const float* x          = (const float*)d_in[0];
  const float* W_in       = (const float*)d_in[1];
  const float* b_in       = (const float*)d_in[2];
  const float* dt_bias    = (const float*)d_in[3];
  const float* conv_k     = (const float*)d_in[4];
  const float* decay_fwd  = (const float*)d_in[5];
  const float* decay_bwd  = (const float*)d_in[6];
  const float* memory     = (const float*)d_in[7];
  const float* mem_decay  = (const float*)d_in[8];
  const float* W_mem_gate = (const float*)d_in[9];
  const float* b_mem_gate = (const float*)d_in[10];
  // d_in[11], d_in[12]: W_slot / b_slot unused by reference
  const float* W_slot_bwd = (const float*)d_in[13];
  const float* b_slot_bwd = (const float*)d_in[14];
  const float* fusion_w   = (const float*)d_in[15];
  const float* scaler_w   = (const float*)d_in[16];
  const float* scaler_b   = (const float*)d_in[17];
  const float* W_out      = (const float*)d_in[18];
  const float* b_out      = (const float*)d_in[19];

  char* ws = (char*)d_ws;
  float*  hbuf    = (float*)(ws + OFF_H);
  float*  fwdbuf  = (float*)(ws + OFF_FWD);
  float*  ybuf    = (float*)(ws + OFF_Y);
  float*  memoutb = (float*)(ws + OFF_MEMOUT);
  bf16_t* xb      = (bf16_t*)(ws + OFF_XB);
  bf16_t* weightb = (bf16_t*)(ws + OFF_WEIGHTB);
  bf16_t* fwdb    = (bf16_t*)(ws + OFF_FWDB);
  bf16_t* WibT    = (bf16_t*)(ws + OFF_WIB);
  float*  logits  = (float*)(ws + OFF_LOGITS);
  bf16_t* swb     = (bf16_t*)(ws + OFF_SWB);
  bf16_t* WslotbT = (bf16_t*)(ws + OFF_WSLOTB);
  bf16_t* membankT= (bf16_t*)(ws + OFF_MEMBANK);
  bf16_t* WoutbT  = (bf16_t*)(ws + OFF_WOUTB);
  float*  gate    = (float*)(ws + OFF_GATE);
  float*  gmem    = (float*)(ws + OFF_GMEM);
  float*  out     = (float*)d_out;

  // 1) conversions (A row-major; all weights transposed to [N][K] bf16)
  {
    size_t nx = (size_t)NTOK * DIM_DI;
    k_cvt_bf16<<<(unsigned)((nx + 255) / 256), 256, 0, stream>>>(x, xb, nx);
    dim3 gWin(TWO_H / 32, DIM_DI / 32);
    k_cvt_bf16_t<<<gWin, 256, 0, stream>>>(W_in, WibT, DIM_DI, TWO_H);
    dim3 gWs(DIM_S / 32, DIM_H / 32);
    k_cvt_bf16_t<<<gWs, 256, 0, stream>>>(W_slot_bwd, WslotbT, DIM_H, DIM_S);
    dim3 gWo(DIM_O / 32, DIM_H / 32);
    k_cvt_bf16_t<<<gWo, 256, 0, stream>>>(W_out, WoutbT, DIM_H, DIM_O);
    dim3 gMb(DIM_H / 32, DIM_S / 32);
    k_membank_t<<<gMb, 256, 0, stream>>>(memory, mem_decay, membankT);
  }

  // 2) in-projection GEMM: h = x @ W_in + b_in   [8192,2048]
  {
    dim3 grid(TWO_H / BN, NTOK / BM);
    k_gemm_bf16_tn<<<grid, 256, 0, stream>>>(xb, WibT, b_in, hbuf, NTOK, TWO_H, DIM_DI);
  }

  // 3) torsion split + conv + silu -> y
  k_torsion_conv<<<(NTOK * DIM_H) / 256, 256, 0, stream>>>(hbuf, dt_bias, conv_k, ybuf);

  // 4) forward EMA scan -> fwd (f32, overwrites h region) + fwd bf16
  k_scan_fwd<<<(DIM_B * DIM_H) / 256, 256, 0, stream>>>(ybuf, decay_fwd, fwdbuf, fwdb);

  // 5) entropy gate + memory gate
  k_entropy_gates<<<NTOK, 256, 0, stream>>>(fwdbuf, W_mem_gate, b_mem_gate,
                                            scaler_w, scaler_b, gate, gmem);

  // 6) slot logits GEMM: fwd @ W_slot_bwd + b_slot_bwd  [8192,128]
  {
    dim3 grid(DIM_S / BN, NTOK / BM);
    k_gemm_bf16_tn<<<grid, 256, 0, stream>>>(fwdb, WslotbT, b_slot_bwd, logits,
                                             NTOK, DIM_S, DIM_H);
  }

  // 7) softmax over slots -> sw (bf16)
  k_softmax128<<<NTOK / 8, 256, 0, stream>>>(logits, swb);

  // 8) memory read GEMM: sw @ mem_bank  [8192,1024]  (gates applied in fusion)
  {
    dim3 grid(DIM_H / BN, NTOK / BM);
    k_gemm_bf16_tn<<<grid, 256, 0, stream>>>(swb, membankT, nullptr, memoutb,
                                             NTOK, DIM_H, DIM_S);
  }

  // 9) reverse scan + contrast + fusion -> weighted (bf16)
  k_scan_bwd_fuse<<<(DIM_B * DIM_H) / 256, 256, 0, stream>>>(
      fwdbuf, memoutb, gate, gmem, decay_bwd, fusion_w, weightb);

  // 10) out projection GEMM: weighted @ W_out + b_out -> d_out (f32)
  {
    dim3 grid(DIM_O / BN, NTOK / BM);
    k_gemm_bf16_tn<<<grid, 256, 0, stream>>>(weightb, WoutbT, b_out, out,
                                             NTOK, DIM_O, DIM_H);
  }
}
